// EncoderLayer_79559974191141
// MI455X (gfx1250) — matrix operand order repaired and checked by reading, not compiled, not run on hardware
//
#include <hip/hip_runtime.h>

// ---------------------------------------------------------------------------
// Bidirectional Mamba encoder layer for MI455X (gfx1250, wave32, WMMA bf16)
// - GEMMs: bf16 WMMA 16x16x32, fp32 accumulate, pre-transposed weights (B^T)
// - Tile staging via GLOBAL_LOAD_ASYNC_TO_LDS_B128 when available
// ---------------------------------------------------------------------------

#define D_MODEL 512
#define D_INNER 1024
#define D_STATE 16
#define D_CONV  4
#define DT_RANK 32
#define D_FF    2048
#define BSZ     4
#define SEQ     2048
#define MROWS   (BSZ * SEQ)   // 8192

typedef __attribute__((ext_vector_type(16))) __bf16 v16bf;
typedef __attribute__((ext_vector_type(8)))  float  v8f;
typedef __attribute__((ext_vector_type(4)))  int    v4i;

union AFrag { unsigned int w[8]; v16bf v; };
union BFrag { uint4 q[2];        v16bf v; };

#if defined(__has_builtin)
#if __has_builtin(__builtin_amdgcn_global_load_async_to_lds_b128)
#define HAVE_ASYNC_LDS 1
#endif
#endif

#ifdef HAVE_ASYNC_LDS
typedef __attribute__((address_space(1))) v4i* gptr_v4i;
typedef __attribute__((address_space(3))) v4i* lptr_v4i;

__device__ __forceinline__ gptr_v4i gbl_cast4(const void* p) {
  // generic and global share address values on AMDGPU
  return (gptr_v4i)(unsigned long long)p;
}
__device__ __forceinline__ lptr_v4i lds_cast4(void* p) {
  // generic LDS pointer's low 32 bits are the LDS offset (aperture mapping)
  return (lptr_v4i)(unsigned)(unsigned long long)p;
}
__device__ __forceinline__ void wait_asynccnt0() {
#if __has_builtin(__builtin_amdgcn_s_wait_asynccnt)
  __builtin_amdgcn_s_wait_asynccnt(0);
#else
  asm volatile("s_wait_asynccnt 0x0" ::: "memory");
#endif
}
#endif

__device__ __forceinline__ unsigned short f2bf(float f) {
  union { float f; unsigned u; } c; c.f = f;
  unsigned u = c.u;
  unsigned r = u + 0x7FFFu + ((u >> 16) & 1u);   // round-to-nearest-even
  return (unsigned short)(r >> 16);
}
__device__ __forceinline__ float bf2f(unsigned short h) {
  union { unsigned u; float f; } c; c.u = ((unsigned)h) << 16;
  return c.f;
}
__device__ __forceinline__ float siluf(float x)     { return x / (1.f + __expf(-x)); }
__device__ __forceinline__ float softplusf(float x) { return x > 20.f ? x : log1pf(__expf(x)); }
__device__ __forceinline__ float geluf(float x) {
  float x3 = x * x * x;
  return 0.5f * x * (1.f + tanhf(0.7978845608028654f * (x + 0.044715f * x3)));
}

// ---------------------------------------------------------------------------
// fp32 (KxN) -> bf16 transposed (NxK) weight convert with placement offsets
// dst[(n + noff) * dld + koff + k] = src[k * N + n]
// ---------------------------------------------------------------------------
__global__ __launch_bounds__(256)
void cvtT_kernel(const float* __restrict__ src, unsigned short* __restrict__ dst,
                 int K, int N, int dld, int noff, int koff) {
  long idx = (long)blockIdx.x * 256 + threadIdx.x;
  long total = (long)K * N;
  if (idx >= total) return;
  int k = (int)(idx / N), n = (int)(idx % N);
  dst[(long)(n + noff) * dld + koff + k] = f2bf(src[idx]);
}

// ---------------------------------------------------------------------------
// LayerNorm over D_MODEL=512, optional keep mask, bf16 output
// ---------------------------------------------------------------------------
__global__ __launch_bounds__(256)
void ln_kernel(const float* __restrict__ x, const float* __restrict__ g,
               const float* __restrict__ bta, const int* __restrict__ mask,
               unsigned short* __restrict__ outH) {
  const long row = blockIdx.x;
  const int tid = threadIdx.x;
  __shared__ float red[256];
  float v0 = x[row * D_MODEL + tid];
  float v1 = x[row * D_MODEL + 256 + tid];
  red[tid] = v0 + v1;
  __syncthreads();
  for (int s = 128; s > 0; s >>= 1) { if (tid < s) red[tid] += red[tid + s]; __syncthreads(); }
  float mean = red[0] * (1.f / 512.f);
  __syncthreads();
  float d0 = v0 - mean, d1 = v1 - mean;
  red[tid] = d0 * d0 + d1 * d1;
  __syncthreads();
  for (int s = 128; s > 0; s >>= 1) { if (tid < s) red[tid] += red[tid + s]; __syncthreads(); }
  float rs = rsqrtf(red[0] * (1.f / 512.f) + 1e-5f);
  float keep = mask ? (1.f - (float)mask[row]) : 1.f;
  outH[row * D_MODEL + tid]       = f2bf((d0 * rs * g[tid]       + bta[tid])       * keep);
  outH[row * D_MODEL + 256 + tid] = f2bf((d1 * rs * g[tid + 256] + bta[tid + 256]) * keep);
}

// ---------------------------------------------------------------------------
// bf16 WMMA GEMM: C[MxN] = act(A[MxK] * B[KxN] + bias) (then residual + C*keep)
// B is passed TRANSPOSED (Bt: N x K, row stride ldbt).
// block tile 128x64, 8 waves of 32x32 (2x2 WMMA 16x16x32), K-step 32
// ---------------------------------------------------------------------------
#define TILE_M 128
#define TILE_N 64
#define TILE_K 32
#define ASTRIDE 40
#define BSTRIDE 40

__global__ __launch_bounds__(256)
void gemm_bf16_kernel(const unsigned short* __restrict__ A, int lda,
                      const unsigned short* __restrict__ Bt, int ldbt,
                      float* __restrict__ outF, unsigned short* __restrict__ outH, int ldo,
                      const float* __restrict__ bias,
                      const float* __restrict__ residual,
                      const int* __restrict__ mask,
                      int K, int act) {
  __shared__ unsigned short As[TILE_M * ASTRIDE];
  __shared__ unsigned short Bs[TILE_N * BSTRIDE];

  const int tid  = threadIdx.x;
  const int wave = tid >> 5;
  const int lane = tid & 31;
  const int wM = wave & 3;      // 4 waves along M (32 rows each)
  const int wN = wave >> 2;     // 2 waves along N (32 cols each)
  const int hf = lane >> 4;     // lane half (wave32 WMMA layout)
  const int r  = lane & 15;
  const long blockM = (long)blockIdx.y * TILE_M;
  const long blockN = (long)blockIdx.x * TILE_N;

  v8f acc[2][2];
  {
    v8f z = {};
    acc[0][0] = z; acc[0][1] = z; acc[1][0] = z; acc[1][1] = z;
  }

  // staging coordinates (8 bf16 = 16B per chunk)
  const int arow0 = tid >> 2;              // A chunk 0 row  (0..63)
  const int arow1 = (tid + 256) >> 2;      // A chunk 1 row  (64..127)
  const int akc   = (tid & 3) << 3;        // A k offset
  const int bn    = tid >> 2;              // B row (n)      (0..63)
  const int bkc   = (tid & 3) << 3;        // B k offset

  for (int k0 = 0; k0 < K; k0 += TILE_K) {
    __syncthreads();
#ifdef HAVE_ASYNC_LDS
    __builtin_amdgcn_global_load_async_to_lds_b128(
        gbl_cast4(A + (blockM + arow0) * lda + k0 + akc),
        lds_cast4(&As[arow0 * ASTRIDE + akc]), 0, 0);
    __builtin_amdgcn_global_load_async_to_lds_b128(
        gbl_cast4(A + (blockM + arow1) * lda + k0 + akc),
        lds_cast4(&As[arow1 * ASTRIDE + akc]), 0, 0);
    __builtin_amdgcn_global_load_async_to_lds_b128(
        gbl_cast4(Bt + (blockN + bn) * ldbt + k0 + bkc),
        lds_cast4(&Bs[bn * BSTRIDE + bkc]), 0, 0);
    wait_asynccnt0();
#else
    {
      uint4 a0 = *(const uint4*)(A + (blockM + arow0) * lda + k0 + akc);
      uint4 a1 = *(const uint4*)(A + (blockM + arow1) * lda + k0 + akc);
      uint4 bv = *(const uint4*)(Bt + (blockN + bn) * ldbt + k0 + bkc);
      *(uint4*)&As[arow0 * ASTRIDE + akc] = a0;
      *(uint4*)&As[arow1 * ASTRIDE + akc] = a1;
      *(uint4*)&Bs[bn * BSTRIDE + bkc]    = bv;
    }
#endif
    __syncthreads();

    // ---- build fragments (16-bit A 16x32 / B 32x16 wave32 layouts)
    AFrag a[2]; BFrag b[2];
#pragma unroll
    for (int mt = 0; mt < 2; mt++) {
      int row = wM * 32 + mt * 16 + r;
#pragma unroll
      for (int e = 0; e < 8; e++) {
        int kk = hf * 16 + (e << 1);  // K pair base
        a[mt].w[e] = *(const unsigned int*)&As[row * ASTRIDE + kk];
      }
    }
#pragma unroll
    for (int nt = 0; nt < 2; nt++) {
      int col = wN * 32 + nt * 16 + r;
      const uint4* p = (const uint4*)&Bs[col * BSTRIDE + hf * 16];
      b[nt].q[0] = p[0];
      b[nt].q[1] = p[1];
    }
#pragma unroll
    for (int mt = 0; mt < 2; mt++)
#pragma unroll
      for (int nt = 0; nt < 2; nt++)
        acc[mt][nt] = __builtin_amdgcn_wmma_f32_16x16x32_bf16(
            false, a[mt].v, false, b[nt].v, (short)0, acc[mt][nt], false, false);
  }

  // ---- epilogue
#pragma unroll
  for (int mt = 0; mt < 2; mt++) {
#pragma unroll
    for (int nt = 0; nt < 2; nt++) {
      union { v8f v; float f[8]; } u; u.v = acc[mt][nt];
      long n = blockN + wN * 32 + nt * 16 + r;
      float bv = bias ? bias[n] : 0.f;
#pragma unroll
      for (int j = 0; j < 8; j++) {
        long m = blockM + wM * 32 + mt * 16 + hf * 8 + j;
        float v = u.f[j] + bv;
        if (act == 1) v = softplusf(v);
        else if (act == 2) v = geluf(v);
        if (residual) {
          float keep = mask ? (1.f - (float)mask[m]) : 1.f;
          v = residual[m * ldo + n] + v * keep;
        }
        long o = m * ldo + n;
        if (outF) outF[o] = v;
        if (outH) outH[o] = f2bf(v);
      }
    }
  }
}

// ---------------------------------------------------------------------------
// depthwise causal conv (K=4) + bias + SiLU. dir=0: causal, dir=1: anti-causal
// (the backward branch's flip/conv/flip folded into original ordering)
// ---------------------------------------------------------------------------
__global__ __launch_bounds__(256)
void conv_silu_kernel(const unsigned short* __restrict__ xz,
                      const float* __restrict__ convw, const float* __restrict__ convb,
                      unsigned short* __restrict__ xc, int dir) {
  long idx = (long)blockIdx.x * 256 + threadIdx.x;   // over MROWS*D_INNER
  int d = (int)(idx & (D_INNER - 1));
  long row = idx >> 10;
  long bb = row >> 11;               // row / SEQ
  int t = (int)(row & (SEQ - 1));
  int off = dir ? 2 * D_INNER : 0;   // xin column block in xz
  float acc = convb[d];
#pragma unroll
  for (int k = 0; k < D_CONV; k++) {
    int tt = dir ? (t + 3 - k) : (t - 3 + k);
    if (tt >= 0 && tt < SEQ)
      acc += convw[d * D_CONV + k] * bf2f(xz[(bb * SEQ + tt) * (4 * D_INNER) + off + d]);
  }
  xc[idx] = f2bf(siluf(acc));
}

// ---------------------------------------------------------------------------
// selective scan: one thread per channel d, 16 fp32 states in registers,
// B_t/C_t staged through LDS; fused output g = (y + x*D) * silu(z)
// ---------------------------------------------------------------------------
__global__ __launch_bounds__(256)
void scan_kernel(const unsigned short* __restrict__ xz,
                 const unsigned short* __restrict__ xc,
                 const unsigned short* __restrict__ dbc,
                 const unsigned short* __restrict__ dt,
                 const float* __restrict__ alog, const float* __restrict__ dparam,
                 unsigned short* __restrict__ G, int dir) {
  const int tid = threadIdx.x;
  const int bb = blockIdx.x >> 2;           // batch
  const int d = (blockIdx.x & 3) * 256 + tid;
  const int zoff = dir ? 3 * D_INNER : D_INNER;
  const int goff = dir ? D_INNER : 0;

  float Arow[D_STATE], h[D_STATE];
#pragma unroll
  for (int n = 0; n < D_STATE; n++) {
    Arow[n] = -__expf(alog[d * D_STATE + n]);
    h[n] = 0.f;
  }
  const float dp = dparam[d];

  __shared__ float sB[D_STATE];
  __shared__ float sC[D_STATE];

  for (int s = 0; s < SEQ; s++) {
    int t = dir ? (SEQ - 1 - s) : s;
    long row = (long)bb * SEQ + t;
    __syncthreads();
    if (tid < 32) {
      float v = bf2f(dbc[row * 64 + DT_RANK + tid]);
      if (tid < D_STATE) sB[tid] = v; else sC[tid - D_STATE] = v;
    }
    __syncthreads();
    float dtv = bf2f(dt[row * D_INNER + d]);
    float xv  = bf2f(xc[row * D_INNER + d]);
    float dtx = dtv * xv;
    float y = 0.f;
#pragma unroll
    for (int n = 0; n < D_STATE; n++) {
      float dA = __expf(dtv * Arow[n]);
      h[n] = dA * h[n] + dtx * sB[n];
      y += h[n] * sC[n];
    }
    float zv = bf2f(xz[row * (4 * D_INNER) + zoff + d]);
    G[row * (2 * D_INNER) + goff + d] = f2bf((y + xv * dp) * siluf(zv));
  }
}

// ---------------------------------------------------------------------------
// host: orchestration
// ---------------------------------------------------------------------------
extern "C" void kernel_launch(void* const* d_in, const int* in_sizes, int n_in,
                              void* d_out, int out_size, void* d_ws, size_t ws_size,
                              hipStream_t stream) {
  (void)in_sizes; (void)n_in; (void)out_size; (void)ws_size;
  const float* x       = (const float*)d_in[0];
  const int*   pmask   = (const int*)  d_in[1];
  const float* n1g     = (const float*)d_in[2];
  const float* n1b     = (const float*)d_in[3];
  const float* n2g     = (const float*)d_in[4];
  const float* n2b     = (const float*)d_in[5];
  const float* inw_f   = (const float*)d_in[6];
  const float* convw_f = (const float*)d_in[7];
  const float* convb_f = (const float*)d_in[8];
  const float* xpw_f   = (const float*)d_in[9];
  const float* dtw_f   = (const float*)d_in[10];
  const float* dtb_f   = (const float*)d_in[11];
  const float* alog_f  = (const float*)d_in[12];
  const float* dpar_f  = (const float*)d_in[13];
  const float* outw_f  = (const float*)d_in[14];
  const float* inw_b   = (const float*)d_in[15];
  const float* convw_b = (const float*)d_in[16];
  const float* convb_b = (const float*)d_in[17];
  const float* xpw_b   = (const float*)d_in[18];
  const float* dtw_b   = (const float*)d_in[19];
  const float* dtb_b   = (const float*)d_in[20];
  const float* alog_b  = (const float*)d_in[21];
  const float* dpar_b  = (const float*)d_in[22];
  const float* outw_b  = (const float*)d_in[23];
  const float* ffn_w1  = (const float*)d_in[24];
  const float* ffn_b1  = (const float*)d_in[25];
  const float* ffn_w2  = (const float*)d_in[26];
  const float* ffn_b2  = (const float*)d_in[27];
  float* out = (float*)d_out;

  // ---- workspace carve-out
  char* wsb = (char*)d_ws;
  size_t off = 0;
  auto alloc = [&](size_t bytes) -> void* {
    void* p = wsb + off;
    off += (bytes + 255) & ~(size_t)255;
    return p;
  };
  typedef unsigned short us;
  us* hbf  = (us*)alloc((size_t)MROWS * D_MODEL * 2);            // LN1 out bf16
  us* h2bf = (us*)alloc((size_t)MROWS * D_MODEL * 2);            // LN2 out bf16
  us* wInT = (us*)alloc((size_t)4 * D_INNER * D_MODEL * 2);      // (4096 x 512)
  us* wXpF = (us*)alloc((size_t)64 * D_INNER * 2);               // (64 x 1024)
  us* wXpB = (us*)alloc((size_t)64 * D_INNER * 2);
  us* wDtF = (us*)alloc((size_t)D_INNER * DT_RANK * 2);          // (1024 x 32)
  us* wDtB = (us*)alloc((size_t)D_INNER * DT_RANK * 2);
  us* wOutT= (us*)alloc((size_t)D_MODEL * 2 * D_INNER * 2);      // (512 x 2048)
  us* wF1T = (us*)alloc((size_t)D_FF * D_MODEL * 2);             // (2048 x 512)
  us* wF2T = (us*)alloc((size_t)D_MODEL * D_FF * 2);             // (512 x 2048)
  us* xz   = (us*)alloc((size_t)MROWS * 4 * D_INNER * 2);        // [xin_f|z_f|xin_b|z_b]
  us* xcF  = (us*)alloc((size_t)MROWS * D_INNER * 2);
  us* xcB  = (us*)alloc((size_t)MROWS * D_INNER * 2);
  us* dbcF = (us*)alloc((size_t)MROWS * 64 * 2);
  us* dbcB = (us*)alloc((size_t)MROWS * 64 * 2);
  us* dtF  = (us*)alloc((size_t)MROWS * D_INNER * 2);
  us* dtB  = (us*)alloc((size_t)MROWS * D_INNER * 2);
  us* G    = (us*)alloc((size_t)MROWS * 2 * D_INNER * 2);        // [g_f | g_b]
  float* x2 = (float*)alloc((size_t)MROWS * D_MODEL * 4);
  us* f1   = (us*)alloc((size_t)MROWS * D_FF * 2);

  auto cvtT = [&](const float* s, us* d, int K, int N, int dld, int noff, int koff) {
    long tot = (long)K * N;
    cvtT_kernel<<<(unsigned)((tot + 255) / 256), 256, 0, stream>>>(s, d, K, N, dld, noff, koff);
  };

  // ---- weight conversion to transposed bf16 (N x K) with packing
  cvtT(inw_f, wInT, D_MODEL, 2 * D_INNER, D_MODEL, 0, 0);
  cvtT(inw_b, wInT, D_MODEL, 2 * D_INNER, D_MODEL, 2 * D_INNER, 0);
  cvtT(xpw_f, wXpF, D_INNER, 64, D_INNER, 0, 0);
  cvtT(xpw_b, wXpB, D_INNER, 64, D_INNER, 0, 0);
  cvtT(dtw_f, wDtF, DT_RANK, D_INNER, DT_RANK, 0, 0);
  cvtT(dtw_b, wDtB, DT_RANK, D_INNER, DT_RANK, 0, 0);
  cvtT(outw_f, wOutT, D_INNER, D_MODEL, 2 * D_INNER, 0, 0);        // K cols 0..1023
  cvtT(outw_b, wOutT, D_INNER, D_MODEL, 2 * D_INNER, 0, D_INNER);  // K cols 1024..2047
  cvtT(ffn_w1, wF1T, D_MODEL, D_FF, D_MODEL, 0, 0);
  cvtT(ffn_w2, wF2T, D_FF, D_MODEL, D_FF, 0, 0);

  dim3 blk(256);
  // ---- LN1 (+keep) -> h bf16
  ln_kernel<<<MROWS, blk, 0, stream>>>(x, n1g, n1b, pmask, hbf);

  // ---- in-proj (both directions fused): xz = h @ [Win_f|Win_b]  (K=512,N=4096)
  gemm_bf16_kernel<<<dim3(4 * D_INNER / TILE_N, MROWS / TILE_M), blk, 0, stream>>>(
      hbf, D_MODEL, wInT, D_MODEL, nullptr, xz, 4 * D_INNER,
      nullptr, nullptr, nullptr, D_MODEL, 0);

  // ---- depthwise conv + SiLU
  conv_silu_kernel<<<(unsigned)((long)MROWS * D_INNER / 256), blk, 0, stream>>>(
      xz, convw_f, convb_f, xcF, 0);
  conv_silu_kernel<<<(unsigned)((long)MROWS * D_INNER / 256), blk, 0, stream>>>(
      xz, convw_b, convb_b, xcB, 1);

  // ---- x-proj: dbc = xc @ xpw  (K=1024, N=64)
  gemm_bf16_kernel<<<dim3(64 / TILE_N, MROWS / TILE_M), blk, 0, stream>>>(
      xcF, D_INNER, wXpF, D_INNER, nullptr, dbcF, 64, nullptr, nullptr, nullptr, D_INNER, 0);
  gemm_bf16_kernel<<<dim3(64 / TILE_N, MROWS / TILE_M), blk, 0, stream>>>(
      xcB, D_INNER, wXpB, D_INNER, nullptr, dbcB, 64, nullptr, nullptr, nullptr, D_INNER, 0);

  // ---- dt = softplus(dt_r @ dtw + dtb)  (K=32, N=1024; A = first 32 cols of dbc)
  gemm_bf16_kernel<<<dim3(D_INNER / TILE_N, MROWS / TILE_M), blk, 0, stream>>>(
      dbcF, 64, wDtF, DT_RANK, nullptr, dtF, D_INNER, dtb_f, nullptr, nullptr, DT_RANK, 1);
  gemm_bf16_kernel<<<dim3(D_INNER / TILE_N, MROWS / TILE_M), blk, 0, stream>>>(
      dbcB, 64, wDtB, DT_RANK, nullptr, dtB, D_INNER, dtb_b, nullptr, nullptr, DT_RANK, 1);

  // ---- selective scans (fwd & bwd), fused gate -> G
  scan_kernel<<<BSZ * (D_INNER / 256), blk, 0, stream>>>(
      xz, xcF, dbcF, dtF, alog_f, dpar_f, G, 0);
  scan_kernel<<<BSZ * (D_INNER / 256), blk, 0, stream>>>(
      xz, xcB, dbcB, dtB, alog_b, dpar_b, G, 1);

  // ---- out-proj (both directions fused): x2 = x + (G @ [Wout_f;Wout_b]) * keep
  gemm_bf16_kernel<<<dim3(D_MODEL / TILE_N, MROWS / TILE_M), blk, 0, stream>>>(
      G, 2 * D_INNER, wOutT, 2 * D_INNER, x2, nullptr, D_MODEL,
      nullptr, x, pmask, 2 * D_INNER, 0);

  // ---- LN2 -> h2 bf16
  ln_kernel<<<MROWS, blk, 0, stream>>>(x2, n2g, n2b, nullptr, h2bf);

  // ---- FFN1: f1 = gelu(h2 @ w1 + b1)
  gemm_bf16_kernel<<<dim3(D_FF / TILE_N, MROWS / TILE_M), blk, 0, stream>>>(
      h2bf, D_MODEL, wF1T, D_MODEL, nullptr, f1, D_FF, ffn_b1, nullptr, nullptr, D_MODEL, 2);

  // ---- FFN2 + residual: out = x2 + f1 @ w2 + b2
  gemm_bf16_kernel<<<dim3(D_MODEL / TILE_N, MROWS / TILE_M), blk, 0, stream>>>(
      f1, D_FF, wF2T, D_FF, out, nullptr, D_MODEL, ffn_b2, x2, nullptr, D_FF, 0);
}
